// Block_81140522156549
// MI455X (gfx1250) — matrix operand order repaired and checked by reading, not compiled, not run on hardware
//
#include <hip/hip_runtime.h>
#include <stdint.h>

#define NE 1024
#define NH 16
#define DH 64
#define BB 2
#define TT 2048
#define MM (BB*TT)

typedef __attribute__((ext_vector_type(16))) __bf16 v16bf;
typedef __attribute__((ext_vector_type(8)))  float  v8f;

union FragBF { v16bf v; uint4 u[2]; };

__device__ __forceinline__ unsigned short f2bf(float f) {
    uint32_t u = __float_as_uint(f);
    u += 0x7FFFu + ((u >> 16) & 1u);           // round-to-nearest-even
    return (unsigned short)(u >> 16);
}

__device__ __forceinline__ v8f vzero() {
    v8f z;
#pragma unroll
    for (int i = 0; i < 8; ++i) z[i] = 0.0f;
    return z;
}

__device__ __forceinline__ v8f wmma_bf16(const FragBF& a, const FragBF& b, v8f c) {
    return __builtin_amdgcn_wmma_f32_16x16x32_bf16(
        false, a.v, false, b.v, (short)0, c, false, false);
}

__device__ __forceinline__ float gelu_tanh(float x) {
    float x3 = x * x * x;
    return 0.5f * x * (1.0f + tanhf(0.7978845608028654f * (x + 0.044715f * x3)));
}

// ---------------- LayerNorm: fp32 in -> bf16 out (one row of 1024 per block) --
__global__ __launch_bounds__(256) void ln_kernel(const float* __restrict__ x,
                                                 const float* __restrict__ w,
                                                 const float* __restrict__ b,
                                                 unsigned short* __restrict__ out) {
    int row = blockIdx.x;
    int tid = threadIdx.x;
    const float* xr = x + (size_t)row * NE;
    float v[4]; float s = 0.f, sq = 0.f;
#pragma unroll
    for (int i = 0; i < 4; ++i) { v[i] = xr[tid + i * 256]; s += v[i]; sq += v[i] * v[i]; }
    __shared__ float rs_[256], rq_[256];
    rs_[tid] = s; rq_[tid] = sq;
    __syncthreads();
    for (int off = 128; off > 0; off >>= 1) {
        if (tid < off) { rs_[tid] += rs_[tid + off]; rq_[tid] += rq_[tid + off]; }
        __syncthreads();
    }
    float mu  = rs_[0] * (1.0f / NE);
    float var = rq_[0] * (1.0f / NE) - mu * mu;
    float inv = rsqrtf(var + 1e-5f);
#pragma unroll
    for (int i = 0; i < 4; ++i) {
        int c = tid + i * 256;
        out[(size_t)row * NE + c] = f2bf((v[i] - mu) * inv * w[c] + b[c]);
    }
}

// ------------- Weight convert + transpose: fp32 [K,N] -> bf16 [N,K] ----------
__global__ __launch_bounds__(256) void wcvt_kernel(const float* __restrict__ W,
                                                   unsigned short* __restrict__ Wt,
                                                   int K, int N) {
    __shared__ float t[32][33];
    int tx = threadIdx.x, ty = threadIdx.y;
    int n0 = blockIdx.x * 32, k0 = blockIdx.y * 32;
#pragma unroll
    for (int i = 0; i < 4; ++i)
        t[ty + i * 8][tx] = W[(size_t)(k0 + ty + i * 8) * N + (n0 + tx)];
    __syncthreads();
#pragma unroll
    for (int i = 0; i < 4; ++i)
        Wt[(size_t)(n0 + ty + i * 8) * K + (k0 + tx)] = f2bf(t[tx][ty + i * 8]);
}

// ---------------- WMMA GEMM: A[M,K] bf16, Bt[N,K] bf16 -> out[M,N] -----------
// Each wave: 32x64 tile (8 wmma accumulators). Block 256T = 8 waves -> 128x128.
template <int EPI /*0=none,1=gelu*/, bool OUTBF>
__global__ __launch_bounds__(256) void gemm_kernel(
        const unsigned short* __restrict__ A, const unsigned short* __restrict__ Bt,
        const float* __restrict__ bias, const float* __restrict__ res,
        void* __restrict__ outp, int M, int N, int K) {
    int lane = threadIdx.x & 31;
    int wave = threadIdx.x >> 5;
    int hf = lane >> 4, ln = lane & 15;
    int m0 = blockIdx.y * 128 + (wave >> 1) * 32;
    int n0 = blockIdx.x * 128 + (wave & 1) * 64;

    v8f acc[2][4];
#pragma unroll
    for (int i = 0; i < 2; ++i)
#pragma unroll
        for (int j = 0; j < 4; ++j) acc[i][j] = vzero();

    const unsigned short* arow[2];
    arow[0] = A + (size_t)(m0 + ln) * K;
    arow[1] = A + (size_t)(m0 + 16 + ln) * K;
    const unsigned short* brow[4];
#pragma unroll
    for (int j = 0; j < 4; ++j) brow[j] = Bt + (size_t)(n0 + j * 16 + ln) * K;

    for (int kb = 0; kb < K; kb += 32) {
        __builtin_prefetch(arow[0] + kb + 128, 0, 1);
        FragBF a[2], b[4];
#pragma unroll
        for (int i = 0; i < 2; ++i) {
            a[i].u[0] = *(const uint4*)(arow[i] + kb + hf * 8);        // k = hf*8 .. +7
            a[i].u[1] = *(const uint4*)(arow[i] + kb + 16 + hf * 8);   // k = 16+hf*8 ..
        }
#pragma unroll
        for (int j = 0; j < 4; ++j) {
            b[j].u[0] = *(const uint4*)(brow[j] + kb + hf * 8);        // k = hf*8 .. +7
            b[j].u[1] = *(const uint4*)(brow[j] + kb + 16 + hf * 8);   // k = 16+hf*8 ..
        }
#pragma unroll
        for (int i = 0; i < 2; ++i)
#pragma unroll
            for (int j = 0; j < 4; ++j)
                acc[i][j] = wmma_bf16(a[i], b[j], acc[i][j]);
    }

#pragma unroll
    for (int i = 0; i < 2; ++i) {
#pragma unroll
        for (int j = 0; j < 4; ++j) {
            int n = n0 + j * 16 + ln;
            float bs = bias[n];
#pragma unroll
            for (int r = 0; r < 8; ++r) {
                int m = m0 + i * 16 + hf * 8 + r;
                float vv = acc[i][j][r] + bs;
                if (EPI == 1) vv = gelu_tanh(vv);
                if (res) vv += res[(size_t)m * N + n];
                if (OUTBF) ((unsigned short*)outp)[(size_t)m * N + n] = f2bf(vv);
                else       ((float*)outp)[(size_t)m * N + n] = vv;
            }
        }
    }
}

// ---------------- Split qkv[B,T,3C] -> Qh/Kh [b,h,t,d], Vt [b,h,d,t] ---------
__global__ __launch_bounds__(256) void split_kernel(const unsigned short* __restrict__ qkv,
                                                    unsigned short* __restrict__ Qh,
                                                    unsigned short* __restrict__ Kh,
                                                    unsigned short* __restrict__ Vt) {
    int e = blockIdx.x * 256 + threadIdx.x;     // [0, MM*NE)
    int bt = e >> 10, c = e & 1023;
    int h = c >> 6, d = c & 63;
    int b = bt >> 11, t = bt & 2047;
    size_t src = (size_t)bt * 3072;
    size_t bh = (size_t)(b * NH + h);
    Qh[(bh * TT + t) * DH + d]                = qkv[src + c];
    Kh[(bh * TT + t) * DH + d]                = qkv[src + 1024 + c];
    Vt[bh * (size_t)DH * TT + (size_t)d * TT + t] = qkv[src + 2048 + c];
}

// ---------------- Flash attention: one wave per 16-row q tile ----------------
__global__ __launch_bounds__(256) void attn_kernel(const unsigned short* __restrict__ Qh,
                                                   const unsigned short* __restrict__ Kh,
                                                   const unsigned short* __restrict__ Vt,
                                                   unsigned short* __restrict__ Y) {
    __shared__ __align__(16) unsigned short pld[8][16][32];
    int lane = threadIdx.x & 31, wave = threadIdx.x >> 5;
    int hf = lane >> 4, ln = lane & 15;
    int qt = blockIdx.x * 8 + wave;              // 0..4095
    int b = qt >> 11;
    int rem = qt & 2047;
    int h = rem >> 7;
    int q0 = (rem & 127) << 4;
    size_t bh = (size_t)(b * NH + h);
    const unsigned short* Qb = Qh + bh * (size_t)TT * DH;
    const unsigned short* Kb = Kh + bh * (size_t)TT * DH;
    const unsigned short* Vb = Vt + bh * (size_t)DH * TT;

    FragBF aQ[2];
    {
        const unsigned short* qr = Qb + (size_t)(q0 + ln) * DH;
        aQ[0].u[0] = *(const uint4*)(qr + hf * 8);
        aQ[0].u[1] = *(const uint4*)(qr + 16 + hf * 8);
        aQ[1].u[0] = *(const uint4*)(qr + 32 + hf * 8);
        aQ[1].u[1] = *(const uint4*)(qr + 48 + hf * 8);
    }

    v8f o[4];
#pragma unroll
    for (int j = 0; j < 4; ++j) o[j] = vzero();
    float mrun[8], lrun[8];
#pragma unroll
    for (int r = 0; r < 8; ++r) { mrun[r] = -1e30f; lrun[r] = 0.0f; }

    int nkb = (q0 + 16 + 31) >> 5;
    for (int ib = 0; ib < nkb; ++ib) {
        int kb = ib * 32;
        v8f s[2];
#pragma unroll
        for (int sub = 0; sub < 2; ++sub) {
            int key = kb + sub * 16 + ln;
            const unsigned short* kr = Kb + (size_t)(key < TT ? key : 0) * DH;
            FragBF b0, b1;
            b0.u[0] = *(const uint4*)(kr + hf * 8);
            b0.u[1] = *(const uint4*)(kr + 16 + hf * 8);
            b1.u[0] = *(const uint4*)(kr + 32 + hf * 8);
            b1.u[1] = *(const uint4*)(kr + 48 + hf * 8);
            v8f sc = vzero();
            sc = wmma_bf16(aQ[0], b0, sc);
            sc = wmma_bf16(aQ[1], b1, sc);
            s[sub] = sc;
        }
        // scale, causal mask, online softmax (row = q0 + hf*8 + r)
#pragma unroll
        for (int r = 0; r < 8; ++r) {
            int qi = q0 + hf * 8 + r;
            float v0 = s[0][r] * 0.125f; if (kb + ln > qi)      v0 = -1e30f;
            float v1 = s[1][r] * 0.125f; if (kb + 16 + ln > qi) v1 = -1e30f;
            float mx = fmaxf(v0, v1);
            mx = fmaxf(mx, __shfl_xor(mx, 1, 32));
            mx = fmaxf(mx, __shfl_xor(mx, 2, 32));
            mx = fmaxf(mx, __shfl_xor(mx, 4, 32));
            mx = fmaxf(mx, __shfl_xor(mx, 8, 32));
            float mnew  = fmaxf(mrun[r], mx);
            float alpha = __expf(mrun[r] - mnew);
            float p0 = __expf(v0 - mnew);
            float p1 = __expf(v1 - mnew);
            float ps = p0 + p1;
            ps += __shfl_xor(ps, 1, 32);
            ps += __shfl_xor(ps, 2, 32);
            ps += __shfl_xor(ps, 4, 32);
            ps += __shfl_xor(ps, 8, 32);
            lrun[r] = lrun[r] * alpha + ps;
            mrun[r] = mnew;
#pragma unroll
            for (int j = 0; j < 4; ++j) o[j][r] *= alpha;
            pld[wave][hf * 8 + r][ln]      = f2bf(p0);
            pld[wave][hf * 8 + r][16 + ln] = f2bf(p1);
        }
        asm volatile("s_wait_dscnt 0" ::: "memory");
        FragBF aP;
        {
            const unsigned short* pr = &pld[wave][ln][0];
            aP.u[0] = *(const uint4*)(pr + hf * 8);
            aP.u[1] = *(const uint4*)(pr + 16 + hf * 8);
        }
        asm volatile("s_wait_dscnt 0" ::: "memory");
#pragma unroll
        for (int j = 0; j < 4; ++j) {
            const unsigned short* vr = Vb + (size_t)(j * 16 + ln) * TT;
            int o0 = kb + hf * 8;      if (o0 > TT - 8) o0 = TT - 8;
            int o1 = kb + 16 + hf * 8; if (o1 > TT - 8) o1 = TT - 8;
            FragBF bv;
            bv.u[0] = *(const uint4*)(vr + o0);
            bv.u[1] = *(const uint4*)(vr + o1);
            o[j] = wmma_bf16(aP, bv, o[j]);
        }
    }
    // epilogue: Y[b,t, h*64 + d] = o / l
#pragma unroll
    for (int r = 0; r < 8; ++r) {
        int t = q0 + hf * 8 + r;
        size_t bt = (size_t)(b * TT + t);
        float inv = 1.0f / lrun[r];
#pragma unroll
        for (int j = 0; j < 4; ++j) {
            int c = h * DH + j * 16 + ln;
            Y[bt * NE + c] = f2bf(o[j][r] * inv);
        }
    }
}

// ---------------------------------------------------------------------------
extern "C" void kernel_launch(void* const* d_in, const int* in_sizes, int n_in,
                              void* d_out, int out_size, void* d_ws, size_t ws_size,
                              hipStream_t stream) {
    const float* x      = (const float*)d_in[0];
    const float* ln1w   = (const float*)d_in[1];
    const float* ln1b   = (const float*)d_in[2];
    const float* W_attn = (const float*)d_in[3];
    const float* b_attn = (const float*)d_in[4];
    const float* W_o    = (const float*)d_in[5];
    const float* b_o    = (const float*)d_in[6];
    const float* ln2w   = (const float*)d_in[7];
    const float* ln2b   = (const float*)d_in[8];
    const float* W_fc   = (const float*)d_in[9];
    const float* b_fc   = (const float*)d_in[10];
    const float* W_proj = (const float*)d_in[11];
    const float* b_proj = (const float*)d_in[12];
    float* out = (float*)d_out;
    (void)in_sizes; (void)n_in; (void)out_size; (void)ws_size;

    char* ws = (char*)d_ws;
    size_t off = 0;
    auto alloc = [&](size_t bytes) -> void* {
        void* p = ws + off;
        off = (off + bytes + 255) & ~(size_t)255;
        return p;
    };
    unsigned short* wt_attn = (unsigned short*)alloc((size_t)3072 * 1024 * 2);
    unsigned short* wt_o    = (unsigned short*)alloc((size_t)1024 * 1024 * 2);
    unsigned short* wt_fc   = (unsigned short*)alloc((size_t)4096 * 1024 * 2);
    unsigned short* wt_proj = (unsigned short*)alloc((size_t)1024 * 4096 * 2);
    unsigned short* hln     = (unsigned short*)alloc((size_t)MM * 1024 * 2);
    unsigned short* qkv     = (unsigned short*)alloc((size_t)MM * 3072 * 2);
    unsigned short* Qhp     = (unsigned short*)alloc((size_t)MM * 1024 * 2);
    unsigned short* Khp     = (unsigned short*)alloc((size_t)MM * 1024 * 2);
    unsigned short* Vtp     = (unsigned short*)alloc((size_t)MM * 1024 * 2);
    unsigned short* yb      = (unsigned short*)alloc((size_t)MM * 1024 * 2);
    float*          x1      = (float*)alloc((size_t)MM * 1024 * 4);
    unsigned short* h2      = (unsigned short*)alloc((size_t)MM * 4096 * 2);

    dim3 tb(32, 8);
    wcvt_kernel<<<dim3(3072 / 32, 1024 / 32), tb, 0, stream>>>(W_attn, wt_attn, 1024, 3072);
    wcvt_kernel<<<dim3(1024 / 32, 1024 / 32), tb, 0, stream>>>(W_o,    wt_o,    1024, 1024);
    wcvt_kernel<<<dim3(4096 / 32, 1024 / 32), tb, 0, stream>>>(W_fc,   wt_fc,   1024, 4096);
    wcvt_kernel<<<dim3(1024 / 32, 4096 / 32), tb, 0, stream>>>(W_proj, wt_proj, 4096, 1024);

    // --- attention path ---
    ln_kernel<<<MM, 256, 0, stream>>>(x, ln1w, ln1b, hln);
    gemm_kernel<0, true><<<dim3(3072 / 128, MM / 128), 256, 0, stream>>>(
        hln, wt_attn, b_attn, nullptr, qkv, MM, 3072, 1024);
    split_kernel<<<(MM * 1024) / 256, 256, 0, stream>>>(qkv, Qhp, Khp, Vtp);
    attn_kernel<<<(MM / 16) / 8, 256, 0, stream>>>(Qhp, Khp, Vtp, yb);
    gemm_kernel<0, false><<<dim3(1024 / 128, MM / 128), 256, 0, stream>>>(
        yb, wt_o, b_o, x, x1, MM, 1024, 1024);            // x1 = x + y@W_o + b_o

    // --- MLP path ---
    ln_kernel<<<MM, 256, 0, stream>>>(x1, ln2w, ln2b, hln);
    gemm_kernel<1, true><<<dim3(4096 / 128, MM / 128), 256, 0, stream>>>(
        hln, wt_fc, b_fc, nullptr, h2, MM, 4096, 1024);   // h2 = gelu(h@W_fc + b_fc)
    gemm_kernel<0, false><<<dim3(1024 / 128, MM / 128), 256, 0, stream>>>(
        h2, wt_proj, b_proj, x1, out, MM, 1024, 4096);    // out = x1 + h2@W_proj + b_proj
}
